// _SelectiveStateSpaceBlock_61203283968398
// MI455X (gfx1250) — matrix operand order repaired and checked by reading, compiled, not run on hardware
//
#include <hip/hip_runtime.h>

// ---------------- types ----------------
typedef __attribute__((ext_vector_type(16))) __bf16 v16bf;
typedef __attribute__((ext_vector_type(8)))  float  v8f;
typedef __attribute__((ext_vector_type(4)))  float  f32x4;
typedef __attribute__((ext_vector_type(4)))  unsigned int v4u;
typedef __attribute__((ext_vector_type(8)))  int v8i;
typedef __attribute__((ext_vector_type(4)))  int v4i;

// LDS (addrspace 3) pointer types -> loads lower to ds_load_* instead of flat_load_*
typedef __attribute__((address_space(3))) const unsigned char* lds_cptr;
typedef __attribute__((address_space(3))) const f32x4*         lds_qptr;

// ---------------- problem constants ----------------
#define BATCH 4
#define SEQ   4096
#define DIM   1024
#define MTOT  (BATCH * SEQ)        // 16384 rows
#define KDIM  1024
#define NTOT  (3 * DIM)            // 3072 fused output cols (a|b|c)
#define NCH   (BATCH * DIM)        // 4096 scan channels
#define NSEG  64
#define SEGLEN (SEQ / NSEG)        // 64

// GEMM tiling
#define KT    64                   // K-step staged per TDM tile
#define NIT   (KDIM / KT)          // 16 iterations
#define ROWB  144                  // LDS row pitch: 128B data + 16B pad (bank-conflict-free)
#define TILEB (128 * ROWB)         // 18432 B per tile buffer

// ---------------- workspace layout (bytes) ----------------
#define OFF_XB   ((size_t)0)                                  // bf16 X: 32 MB
#define OFF_WB   (OFF_XB + (size_t)MTOT * KDIM * 2)           // bf16 W packed [3072][1024]: 6 MB
#define OFF_BIAS (OFF_WB + (size_t)NTOT * KDIM * 2)           // f32 bias[3072]
#define OFF_ABC  (OFF_BIAS + (size_t)NTOT * 4)                // f32 activated [16384][3072]: 192 MB
#define OFF_ASEG (OFF_ABC + (size_t)MTOT * NTOT * 4)          // f32 [NSEG][NCH]
#define OFF_BSEG (OFF_ASEG + (size_t)NSEG * NCH * 4)
#define OFF_SST  (OFF_BSEG + (size_t)NSEG * NCH * 4)

// ---------------- conversion kernels ----------------
__global__ void convert_x_bf16(const float* __restrict__ x, __bf16* __restrict__ xb, int n) {
    int i = blockIdx.x * blockDim.x + threadIdx.x;
    if (i < n) xb[i] = (__bf16)x[i];
}

__global__ void pack_w_bias(const float* __restrict__ Wa, const float* __restrict__ Wb,
                            const float* __restrict__ Wc,
                            const float* __restrict__ ba, const float* __restrict__ bb,
                            const float* __restrict__ bc,
                            __bf16* __restrict__ wout, float* __restrict__ biasout) {
    int i = blockIdx.x * blockDim.x + threadIdx.x;
    const int WN = DIM * KDIM;                  // 2^20
    if (i < 3 * WN) {
        int which = i >> 20;
        int r = i & (WN - 1);
        float v = (which == 0) ? Wa[r] : (which == 1) ? Wb[r] : Wc[r];
        wout[i] = (__bf16)v;
    }
    if (i < NTOT) {
        int which = i >> 10;
        int r = i & (DIM - 1);
        biasout[i] = (which == 0) ? ba[r] : (which == 1) ? bb[r] : bc[r];
    }
}

// ---------------- TDM 2D tile load: global(bf16, row-stride 1024) -> LDS ----------------
// D# per CDNA5 ISA ch.8: group0 {count,lds,global,type}, group1 {flags,dims,tile,stride}.
// pad_enable inserts 4 DWORDs (16B) after every 32 DWORDs (one 64-elem bf16 row = 128B),
// so LDS rows land at pitch 144B -> ds_load_b128 fragment reads are bank-conflict-free.
__device__ __forceinline__ void tdm_load_tile(unsigned lds_off, const __bf16* gptr) {
    unsigned long long ga = (unsigned long long)(uintptr_t)gptr;
    v4u g0;
    g0.x = 1u;                                   // count=1 (valid descriptor)
    g0.y = lds_off;                              // LDS byte address
    g0.z = (unsigned)ga;                         // global_addr[31:0]
    g0.w = (unsigned)(ga >> 32) | 0x80000000u;   // global_addr[56:32] | type=2 @ [127:126]

    const unsigned dw0 = (1u << 16)              // data_size = 2 bytes
                       | (1u << 20)              // pad_enable
                       | (4u << 22)              // pad_interval: every 32 DWORDs
                       | (3u << 25);             // pad_amount: 4 DWORDs (16B)
    const unsigned td0 = (unsigned)KDIM;         // tensor_dim0 (>= tile_dim0, no OOB)
    const unsigned td1 = (unsigned)MTOT;         // tensor_dim1 (>= rows)
    v8i g1;
    g1[0] = (int)dw0;
    g1[1] = (int)((td0 & 0xFFFFu) << 16);                      // [47:32]=0, [63:48]=td0.lo
    g1[2] = (int)((td0 >> 16) | ((td1 & 0xFFFFu) << 16));      // td0.hi | td1.lo
    g1[3] = (int)((td1 >> 16) | ((unsigned)KT << 16));         // td1.hi | tile_dim0=64
    g1[4] = (int)128u;                                         // tile_dim1=128, tile_dim2=0
    g1[5] = (int)(unsigned)KDIM;                               // tensor_dim0_stride lo32 = 1024
    g1[6] = 0;                                                 // stride0 hi | stride1 lo
    g1[7] = 0;                                                 // stride1 hi
    v4i g2 = {1, 1, 0, 0};                                     // tensor_dim2=1, tensor_dim3=1
    v4i g3 = {0, 0, 0, 0};
    v8i g4 = {0, 0, 0, 0, 0, 0, 0, 0};                         // extra group (clang-23 6-arg form)
    __builtin_amdgcn_tensor_load_to_lds(g0, g1, g2, g3, g4, 0);
}

// ---------------- fast activations (no IEEE divide) ----------------
// k==1 -> sigmoid(v); k==2 -> tanh(v) = 2*sigmoid(2v)-1.  act = k*rcp(1+exp(-k v)) - (k-1)
__device__ __forceinline__ float gate_act(float v, float k) {
    float r = __builtin_amdgcn_rcpf(1.0f + __expf(-k * v));
    return __builtin_fmaf(k, r, -(k - 1.0f));
}

// ---------------- fused triple-GEMM + activation epilogue ----------------
// out[m][n] over M=16384, N=3072, K=1024.  Block tile 128x128, 8 waves (2x4),
// wave tile 64x32 = 4x2 WMMA tiles.  A/B tiles staged to LDS by TDM, double-buffered.
__global__ __launch_bounds__(256) void fused_gemm3_wmma(
    const __bf16* __restrict__ X, const __bf16* __restrict__ W,
    const float* __restrict__ bias, float* __restrict__ out)
{
    __shared__ __align__(16) unsigned char smem[4 * TILEB];   // A0 A1 B0 B1

    // Real LDS byte offset of smem: flat LDS addresses carry the LDS offset in addr[31:0].
    const unsigned lds_base = (unsigned)(uintptr_t)(void*)&smem[0];

    const int n0 = blockIdx.x * 128;
    const int m0 = blockIdx.y * 128;
    const int tid  = threadIdx.x;
    const int wave = tid >> 5;
    const int lane = tid & 31;
    const int wm = (wave >> 2) << 6;   // 0 or 64
    const int wn = (wave & 3) << 5;    // 0,32,64,96
    const int r  = lane & 15;
    const int hi = lane >> 4;

    const __bf16* Ag = X + (size_t)m0 * KDIM;   // block A tile origin (row m0, k 0)
    const __bf16* Bg = W + (size_t)n0 * KDIM;   // block B tile origin (row n0, k 0)

    // prologue: stage tile 0
    if (wave == 0) tdm_load_tile(lds_base,             Ag);
    if (wave == 1) tdm_load_tile(lds_base + 2 * TILEB, Bg);

    // per-lane LDS fragment byte offsets (within a tile buffer)
    // A frag (tile i, ksub kk): row wm+i*16+r, chunks at k=kk+hi*8 and k=kk+16+hi*8
    // B frag (tile j, ksub kk): row wn+j*16+r, chunks at k=kk+hi*8 and k=kk+16+hi*8
    v8f acc[4][2] = {};

    for (int it = 0; it < NIT; ++it) {
        __syncthreads();   // all reads of buffer (it+1)&1 from previous iter finished
        if (wave == 0) {
            if (it + 1 < NIT) {
                tdm_load_tile(lds_base + (unsigned)(((it + 1) & 1) * TILEB), Ag + (it + 1) * KT);
                __builtin_amdgcn_s_wait_tensorcnt(1);
            } else {
                __builtin_amdgcn_s_wait_tensorcnt(0);
            }
        }
        if (wave == 1) {
            if (it + 1 < NIT) {
                tdm_load_tile(lds_base + (unsigned)(2 * TILEB + ((it + 1) & 1) * TILEB),
                              Bg + (it + 1) * KT);
                __builtin_amdgcn_s_wait_tensorcnt(1);
            } else {
                __builtin_amdgcn_s_wait_tensorcnt(0);
            }
        }
        __syncthreads();   // buffer it&1 fully written by TDM

        // Launder the 32-bit LDS byte offsets (TDM writes LDS through an integer
        // descriptor address, so the compiler would otherwise fold the reads of
        // "never-written" smem), then rebuild explicit addrspace(3) pointers so
        // the loads lower to ds_load_b128 (not flat_load).
        unsigned aoff = lds_base + (unsigned)((it & 1) * TILEB);
        unsigned boff = lds_base + (unsigned)(2 * TILEB + (it & 1) * TILEB);
        asm volatile("" : "+v"(aoff), "+v"(boff) :: "memory");
        lds_cptr As = (lds_cptr)(uintptr_t)aoff;
        lds_cptr Bs = (lds_cptr)(uintptr_t)boff;

#pragma unroll
        for (int kk = 0; kk < KT; kk += 32) {
            v16bf a[4], b[2];
#pragma unroll
            for (int i = 0; i < 4; ++i) {
                lds_cptr p = As + (wm + i * 16 + r) * ROWB + (kk + hi * 8) * 2;
                union { f32x4 q[2]; v16bf v; } u;
                u.q[0] = *(lds_qptr)(p);
                u.q[1] = *(lds_qptr)(p + 32);   // k + 16
                a[i] = u.v;
            }
#pragma unroll
            for (int j = 0; j < 2; ++j) {
                lds_cptr p = Bs + (wn + j * 16 + r) * ROWB + (kk + hi * 8) * 2;
                union { f32x4 q[2]; v16bf v; } u;
                u.q[0] = *(lds_qptr)(p);
                u.q[1] = *(lds_qptr)(p + 32);   // k + 16
                b[j] = u.v;
            }
#pragma unroll
            for (int i = 0; i < 4; ++i)
#pragma unroll
                for (int j = 0; j < 2; ++j)
                    acc[i][j] = __builtin_amdgcn_wmma_f32_16x16x32_bf16(
                        false, a[i], false, b[j], (short)0, acc[i][j], false, false);
        }
    }

    // Epilogue: + bias, branchless sigmoid/tanh/sigmoid by segment, store fp32
    const int nl = r;          // n within 16-tile
    const int mb = hi * 8;     // m base within 16-tile
#pragma unroll
    for (int i = 0; i < 4; ++i) {
#pragma unroll
        for (int j = 0; j < 2; ++j) {
            const int ng = n0 + wn + j * 16 + nl;
            const float kgate = ((ng >> 10) == 1) ? 2.0f : 1.0f;   // tanh for middle third
            const float bs = bias[ng];
#pragma unroll
            for (int rr = 0; rr < 8; ++rr) {
                float v = acc[i][j][rr] + bs;
                const int gm = m0 + wm + i * 16 + mb + rr;
                out[(size_t)gm * NTOT + ng] = gate_act(v, kgate);
            }
        }
    }
}

// ---------------- chunked parallel scan ----------------
__global__ void scan_phase1(const float* __restrict__ abc,
                            float* __restrict__ Aseg, float* __restrict__ Bseg) {
    int tid = blockIdx.x * blockDim.x + threadIdx.x;   // 0 .. NCH*NSEG-1
    int ch  = tid & (NCH - 1);
    int seg = tid >> 12;
    int b = ch >> 10, d = ch & (DIM - 1);
    size_t row = ((size_t)b * SEQ + (size_t)seg * SEGLEN) * NTOT + d;
    float A = 1.0f, Bv = 0.0f;
    for (int t = 0; t < SEGLEN; ++t) {
        float av = abc[row];
        float bv = abc[row + DIM];
        Bv = av * Bv + (1.0f - av) * bv;
        A *= av;
        row += NTOT;
    }
    Aseg[(size_t)seg * NCH + ch] = A;
    Bseg[(size_t)seg * NCH + ch] = Bv;
}

__global__ void scan_phase2(const float* __restrict__ Aseg, const float* __restrict__ Bseg,
                            float* __restrict__ Sstart) {
    int ch = blockIdx.x * blockDim.x + threadIdx.x;    // 0..NCH-1
    float s = 0.0f;
    for (int seg = 0; seg < NSEG; ++seg) {
        Sstart[(size_t)seg * NCH + ch] = s;
        s = Aseg[(size_t)seg * NCH + ch] * s + Bseg[(size_t)seg * NCH + ch];
    }
}

__global__ void scan_phase3(const float* __restrict__ abc, const float* __restrict__ Sstart,
                            float* __restrict__ y) {
    int tid = blockIdx.x * blockDim.x + threadIdx.x;
    int ch  = tid & (NCH - 1);
    int seg = tid >> 12;
    int b = ch >> 10, d = ch & (DIM - 1);
    size_t row  = ((size_t)b * SEQ + (size_t)seg * SEGLEN) * NTOT + d;
    size_t yidx = ((size_t)b * SEQ + (size_t)seg * SEGLEN) * DIM + d;
    float s = Sstart[(size_t)seg * NCH + ch];
    for (int t = 0; t < SEGLEN; ++t) {
        float av = abc[row];
        float bv = abc[row + DIM];
        float cv = abc[row + 2 * DIM];
        s = av * s + (1.0f - av) * bv;
        y[yidx] = cv * s;
        row  += NTOT;
        yidx += DIM;
    }
}

// ---------------- host entry ----------------
extern "C" void kernel_launch(void* const* d_in, const int* in_sizes, int n_in,
                              void* d_out, int out_size, void* d_ws, size_t ws_size,
                              hipStream_t stream) {
    (void)in_sizes; (void)n_in; (void)out_size; (void)ws_size;
    const float* X  = (const float*)d_in[0];
    const float* Wa = (const float*)d_in[1];
    const float* ba = (const float*)d_in[2];
    const float* Wb = (const float*)d_in[3];
    const float* bb = (const float*)d_in[4];
    const float* Wc = (const float*)d_in[5];
    const float* bc = (const float*)d_in[6];
    float* y = (float*)d_out;

    char* ws = (char*)d_ws;
    __bf16* XB   = (__bf16*)(ws + OFF_XB);
    __bf16* WB   = (__bf16*)(ws + OFF_WB);
    float*  BIAS = (float*)(ws + OFF_BIAS);
    float*  ABC  = (float*)(ws + OFF_ABC);
    float*  ASEG = (float*)(ws + OFF_ASEG);
    float*  BSEG = (float*)(ws + OFF_BSEG);
    float*  SST  = (float*)(ws + OFF_SST);

    const int nX = MTOT * KDIM;                       // 16.7M
    convert_x_bf16<<<(nX + 255) / 256, 256, 0, stream>>>(X, XB, nX);

    const int nW = 3 * DIM * KDIM;                    // 3.1M
    pack_w_bias<<<(nW + 255) / 256, 256, 0, stream>>>(Wa, Wb, Wc, ba, bb, bc, WB, BIAS);

    dim3 gg(NTOT / 128, MTOT / 128);                  // 24 x 128 blocks
    fused_gemm3_wmma<<<gg, 256, 0, stream>>>(XB, WB, BIAS, ABC);

    scan_phase1<<<(NCH * NSEG) / 256, 256, 0, stream>>>(ABC, ASEG, BSEG);
    scan_phase2<<<NCH / 256, 256, 0, stream>>>(ASEG, BSEG, SST);
    scan_phase3<<<(NCH * NSEG) / 256, 256, 0, stream>>>(ABC, SST, y);
}
